// Decoder_73778948210772
// MI455X (gfx1250) — matrix operand order repaired and checked by reading, not compiled, not run on hardware
//
#include <hip/hip_runtime.h>
#include <hip/hip_bf16.h>
#include <math.h>

#define T_DIM 64
#define S_DIM 400
#define B_DIM 32
#define H_DIM 512
#define E_DIM 1024
#define I_DIM 128

typedef __attribute__((ext_vector_type(16))) __bf16 bf16x16;
typedef __attribute__((ext_vector_type(8)))  float  f32x8;

// ---------------------------------------------------------------------------
// Fragment layouts used here (16x16x32 bf16 WMMA; contiguous k order, same in A and B):
// A 16x32 (MxK): lane m = lane&15, half = lane>>4.
//   One contiguous run of 16 elements at [k0+16*half, +16) along K.
// B 32x16 (KxN): lane n = lane&15, half = lane>>4.
//   One contiguous run of 16 elements at [k0+16*half, +16) along K.
// C/D f32 16x16: n = lane&15, m = 8*(lane>>4) + v
// ---------------------------------------------------------------------------

// A fragment from f32 activations: 4x b128 loads + pack to bf16.
__device__ __forceinline__ bf16x16 load_a_frag(const float* __restrict__ A,
                                               int lda, int row0, int k0) {
  int lane = threadIdx.x & 31;
  int m = lane & 15, half = lane >> 4;
  const float* r0 = A + (size_t)(row0 + m) * lda + k0 + 16 * half;
  float4 p0 = *reinterpret_cast<const float4*>(r0);
  float4 p1 = *reinterpret_cast<const float4*>(r0 + 4);
  float4 p2 = *reinterpret_cast<const float4*>(r0 + 8);
  float4 p3 = *reinterpret_cast<const float4*>(r0 + 12);
  bf16x16 a;
  a[0]  = (__bf16)p0.x; a[1]  = (__bf16)p0.y; a[2]  = (__bf16)p0.z; a[3]  = (__bf16)p0.w;
  a[4]  = (__bf16)p1.x; a[5]  = (__bf16)p1.y; a[6]  = (__bf16)p1.z; a[7]  = (__bf16)p1.w;
  a[8]  = (__bf16)p2.x; a[9]  = (__bf16)p2.y; a[10] = (__bf16)p2.z; a[11] = (__bf16)p2.w;
  a[12] = (__bf16)p3.x; a[13] = (__bf16)p3.y; a[14] = (__bf16)p3.z; a[15] = (__bf16)p3.w;
  return a;
}

// B fragment from pre-converted bf16 weights: single 32-byte vector load.
// W row-major (Nrows x K); B[k][n] = W[n0+n, k0+k].
__device__ __forceinline__ bf16x16 load_b_frag(const __bf16* __restrict__ W,
                                               int ldw, int n0, int k0) {
  int lane = threadIdx.x & 31;
  int n = lane & 15, half = lane >> 4;
  const __bf16* base = W + (size_t)(n0 + n) * ldw + k0 + 16 * half;
  return *reinterpret_cast<const bf16x16*>(base);
}

__device__ __forceinline__ f32x8 wmma_bf16(bf16x16 a, bf16x16 b, f32x8 c) {
  return __builtin_amdgcn_wmma_f32_16x16x32_bf16(false, a, false, b,
                                                 (short)0, c, false, false);
}

__device__ __forceinline__ float sigmoidf_(float x) {
  return 1.0f / (1.0f + __expf(-x));
}

// ---------------------------------------------------------------------------
// One-time f32 -> bf16 weight conversion.
// ---------------------------------------------------------------------------
__global__ void cvt_bf16_kernel(const float* __restrict__ in,
                                __bf16* __restrict__ out, int n) {
  int i = blockIdx.x * 256 + threadIdx.x;
  if (i < n) out[i] = (__bf16)in[i];
}

// ---------------------------------------------------------------------------
// Precompute: Wh_enc[m, f] = sum_e enc[m, e] * W_h[f, e], m in [0, S*B)
// grid(S*B/16, E/64), block 128 (4 waves; each wave one 16x16 tile)
// ---------------------------------------------------------------------------
__global__ void whenc_kernel(const float* __restrict__ enc,
                             const __bf16* __restrict__ W_hb,
                             float* __restrict__ out) {
  int wave = threadIdx.x >> 5;
  int row0 = blockIdx.x * 16;
  int n0 = (blockIdx.y * 4 + wave) * 16;
  f32x8 acc = {};
  for (int k0 = 0; k0 < E_DIM; k0 += 32)
    acc = wmma_bf16(load_a_frag(enc, E_DIM, row0, k0),
                    load_b_frag(W_hb, E_DIM, n0, k0), acc);
  int lane = threadIdx.x & 31;
  int n = lane & 15;
#pragma unroll
  for (int v = 0; v < 8; ++v) {
    int m = row0 + (lane >> 4) * 8 + v;
    out[(size_t)m * E_DIM + n0 + n] = acc[v];
  }
}

// ---------------------------------------------------------------------------
// Fused GRU cell + mask-blend.  out = ym*GRU(x,h) + (1-ym)*h
// grid(H/64), block 128 (4 waves; each wave: 16 cols x both 16-row halves)
// ---------------------------------------------------------------------------
__global__ void gru_kernel(const float* __restrict__ x, int Kx,
                           const float* __restrict__ h,
                           const __bf16* __restrict__ w_ih,
                           const __bf16* __restrict__ w_hh,
                           const float* __restrict__ b_ih,
                           const float* __restrict__ b_hh,
                           const float* __restrict__ ym,
                           float* __restrict__ out,
                           float* __restrict__ out2) {
  const int H = H_DIM;
  int wave = threadIdx.x >> 5;
  int col0 = (blockIdx.x * 4 + wave) * 16;
  int lane = threadIdx.x & 31;
  int n = lane & 15;

  for (int mt = 0; mt < 2; ++mt) {
    int row0 = mt * 16;
    f32x8 accR = {}, accZ = {}, accNi = {}, accNh = {};
    for (int k0 = 0; k0 < Kx; k0 += 32) {
      bf16x16 a = load_a_frag(x, Kx, row0, k0);
      accR  = wmma_bf16(a, load_b_frag(w_ih, Kx, 0 * H + col0, k0), accR);
      accZ  = wmma_bf16(a, load_b_frag(w_ih, Kx, 1 * H + col0, k0), accZ);
      accNi = wmma_bf16(a, load_b_frag(w_ih, Kx, 2 * H + col0, k0), accNi);
    }
    for (int k0 = 0; k0 < H; k0 += 32) {
      bf16x16 a = load_a_frag(h, H, row0, k0);
      accR  = wmma_bf16(a, load_b_frag(w_hh, H, 0 * H + col0, k0), accR);
      accZ  = wmma_bf16(a, load_b_frag(w_hh, H, 1 * H + col0, k0), accZ);
      accNh = wmma_bf16(a, load_b_frag(w_hh, H, 2 * H + col0, k0), accNh);
    }
    float bR  = b_ih[0 * H + col0 + n] + b_hh[0 * H + col0 + n];
    float bZ  = b_ih[1 * H + col0 + n] + b_hh[1 * H + col0 + n];
    float bNi = b_ih[2 * H + col0 + n];
    float bNh = b_hh[2 * H + col0 + n];
#pragma unroll
    for (int v = 0; v < 8; ++v) {
      int m = row0 + (lane >> 4) * 8 + v;
      float hv = h[m * H + col0 + n];
      float r  = sigmoidf_(accR[v] + bR);
      float z  = sigmoidf_(accZ[v] + bZ);
      float nn = tanhf((accNi[v] + bNi) + r * (accNh[v] + bNh));
      float g  = (1.0f - z) * nn + z * hv;
      float mk = ym[m];
      float s  = mk * g + (1.0f - mk) * hv;
      out[m * H + col0 + n] = s;
      if (out2) out2[m * H + col0 + n] = s;
    }
  }
}

// ---------------------------------------------------------------------------
// proj[b, f] = sum_h s1[b,h] * W_s[f,h] + b_attn[f].  grid(E/64), block 128.
// ---------------------------------------------------------------------------
__global__ void proj_kernel(const float* __restrict__ s1,
                            const __bf16* __restrict__ W_sb,
                            const float* __restrict__ b_attn,
                            float* __restrict__ proj) {
  int wave = threadIdx.x >> 5;
  int col0 = (blockIdx.x * 4 + wave) * 16;
  int lane = threadIdx.x & 31;
  int n = lane & 15;
  for (int mt = 0; mt < 2; ++mt) {
    int row0 = mt * 16;
    f32x8 acc = {};
    for (int k0 = 0; k0 < H_DIM; k0 += 32)
      acc = wmma_bf16(load_a_frag(s1, H_DIM, row0, k0),
                      load_b_frag(W_sb, H_DIM, col0, k0), acc);
    float ba = b_attn[col0 + n];
#pragma unroll
    for (int v = 0; v < 8; ++v) {
      int m = row0 + (lane >> 4) * 8 + v;
      proj[m * E_DIM + col0 + n] = acc[v] + ba;
    }
  }
}

// ---------------------------------------------------------------------------
// score[s,b] = xm==0 ? -1e9 : xm * sum_e V[e]*tanh(WhEnc[s,b,e]+proj[b,e]+cov[b,s]*wc[e])
// grid(S, B), block 128.  Streams Wh_enc out of L2.
// ---------------------------------------------------------------------------
__global__ void score_kernel(const float* __restrict__ WhEnc,
                             const float* __restrict__ proj,
                             const float* __restrict__ cov,
                             const float* __restrict__ wc,
                             const float* __restrict__ Vv,
                             const float* __restrict__ xmask,
                             float* __restrict__ score) {
  int s = blockIdx.x, bb = blockIdx.y;
  const float* we = WhEnc + ((size_t)s * B_DIM + bb) * E_DIM;
  const float* pr = proj + (size_t)bb * E_DIM;
  float cv = cov[bb * S_DIM + s];
  float xm = xmask[s * B_DIM + bb];
  float acc = 0.0f;
  for (int e = threadIdx.x; e < E_DIM; e += 128)
    acc += Vv[e] * tanhf(we[e] + pr[e] + cv * wc[e]);
  __shared__ float red[128];
  red[threadIdx.x] = acc;
  __syncthreads();
  for (int off = 64; off > 0; off >>= 1) {
    if (threadIdx.x < off) red[threadIdx.x] += red[threadIdx.x + off];
    __syncthreads();
  }
  if (threadIdx.x == 0)
    score[s * B_DIM + bb] = (xm == 0.0f) ? -1.0e9f : red[0] * xm;
}

// ---------------------------------------------------------------------------
// Softmax over S per batch, emit wa[t], coverage[t] (pre-update), update cov.
// grid(B), block 512.
// ---------------------------------------------------------------------------
__global__ void softmax_kernel(const float* __restrict__ score,
                               float* __restrict__ aw,
                               float* __restrict__ cov,
                               float* __restrict__ out_wa,
                               float* __restrict__ out_cov) {
  int bb = blockIdx.x;
  int tid = threadIdx.x;
  __shared__ float red[512];
  float v = (tid < S_DIM) ? score[tid * B_DIM + bb] : -3.0e38f;
  red[tid] = v;
  __syncthreads();
  for (int off = 256; off > 0; off >>= 1) {
    if (tid < off) red[tid] = fmaxf(red[tid], red[tid + off]);
    __syncthreads();
  }
  float mx = red[0];
  __syncthreads();
  float e = (tid < S_DIM) ? __expf(v - mx) : 0.0f;
  red[tid] = e;
  __syncthreads();
  for (int off = 256; off > 0; off >>= 1) {
    if (tid < off) red[tid] += red[tid + off];
    __syncthreads();
  }
  float sm = red[0];
  if (tid < S_DIM) {
    float a = e / sm;
    aw[tid * B_DIM + bb] = a;
    out_wa[bb * S_DIM + tid] = a;
    float c = cov[bb * S_DIM + tid];
    out_cov[bb * S_DIM + tid] = c;      // coverage BEFORE adding wa
    cov[bb * S_DIM + tid] = c + a;
  }
}

// ---------------------------------------------------------------------------
// ctx[b,e] = sum_s aw[s,b] * enc[s,b,e].  grid(E/256, B), block 256.
// Prefetch ahead along S to keep the L2 stream busy (global_prefetch_b8).
// ---------------------------------------------------------------------------
__global__ void ctx_kernel(const float* __restrict__ aw,
                           const float* __restrict__ enc,
                           float* __restrict__ ctx_ws,
                           float* __restrict__ out_ctx) {
  int bb = blockIdx.y;
  int e = blockIdx.x * 256 + threadIdx.x;
  float acc = 0.0f;
  for (int s = 0; s < S_DIM; ++s) {
    if (s + 8 < S_DIM)
      __builtin_prefetch(&enc[((size_t)(s + 8) * B_DIM + bb) * E_DIM + e], 0, 0);
    acc += aw[s * B_DIM + bb] * enc[((size_t)s * B_DIM + bb) * E_DIM + e];
  }
  ctx_ws[bb * E_DIM + e] = acc;
  out_ctx[bb * E_DIM + e] = acc;
}

// x_indexes[t,b,s] = x_index[s,b] (as float, exact for < 2^24)
__global__ void xidx_kernel(const int* __restrict__ xi, float* __restrict__ o) {
  int idx = blockIdx.x * 256 + threadIdx.x;
  if (idx >= T_DIM * B_DIM * S_DIM) return;
  int s = idx % S_DIM;
  int b = (idx / S_DIM) % B_DIM;
  o[idx] = (float)xi[s * B_DIM + b];
}

__global__ void init_kernel(const float* __restrict__ h0,
                            const float* __restrict__ c0,
                            float* __restrict__ hidden,
                            float* __restrict__ cov) {
  int i = blockIdx.x * 256 + threadIdx.x;
  if (i < B_DIM * H_DIM) hidden[i] = h0[i];
  if (i < B_DIM * S_DIM) cov[i] = c0[i];
}

// ---------------------------------------------------------------------------
extern "C" void kernel_launch(void* const* d_in, const int* in_sizes, int n_in,
                              void* d_out, int out_size, void* d_ws,
                              size_t ws_size, hipStream_t stream) {
  (void)in_sizes; (void)n_in; (void)out_size; (void)ws_size;
  const float* emb   = (const float*)d_in[0];
  const float* enc   = (const float*)d_in[1];
  const float* h0    = (const float*)d_in[2];
  const float* xmask = (const float*)d_in[3];
  const float* ymask = (const float*)d_in[4];
  const int*   xidx  = (const int*)d_in[5];
  const float* cov0  = (const float*)d_in[6];
  const float* W_h   = (const float*)d_in[7];
  const float* W_s   = (const float*)d_in[8];
  const float* W_c   = (const float*)d_in[9];
  const float* battn = (const float*)d_in[10];
  const float* Vv    = (const float*)d_in[11];
  const float* wih1  = (const float*)d_in[12];
  const float* whh1  = (const float*)d_in[13];
  const float* bih1  = (const float*)d_in[14];
  const float* bhh1  = (const float*)d_in[15];
  const float* wih2  = (const float*)d_in[16];
  const float* whh2  = (const float*)d_in[17];
  const float* bih2  = (const float*)d_in[18];
  const float* bhh2  = (const float*)d_in[19];

  float* out = (float*)d_out;
  float* out_hid = out;                                      // T*B*H
  float* out_ctx = out_hid + (size_t)T_DIM * B_DIM * H_DIM;  // T*B*E
  float* out_wa  = out_ctx + (size_t)T_DIM * B_DIM * E_DIM;  // T*B*S
  float* out_xi  = out_wa  + (size_t)T_DIM * B_DIM * S_DIM;  // T*B*S
  float* out_cov = out_xi  + (size_t)T_DIM * B_DIM * S_DIM;  // T*B*S

  // ---- workspace layout (f32 region first, then bf16 weights) ----
  float* W = (float*)d_ws;
  float* whenc  = W; W += (size_t)S_DIM * B_DIM * E_DIM;  // 52.4 MB
  float* hidden = W; W += B_DIM * H_DIM;
  float* s1     = W; W += B_DIM * H_DIM;
  float* proj   = W; W += B_DIM * E_DIM;
  float* score  = W; W += S_DIM * B_DIM;
  float* aw     = W; W += S_DIM * B_DIM;
  float* cov    = W; W += B_DIM * S_DIM;
  float* ctx    = W; W += B_DIM * E_DIM;

  __bf16* Bp = (__bf16*)W;
  const int N_WIH1 = 3 * H_DIM * I_DIM;   // 196608
  const int N_WHH1 = 3 * H_DIM * H_DIM;   // 786432
  const int N_WIH2 = 3 * H_DIM * E_DIM;   // 1572864
  const int N_WHH2 = 3 * H_DIM * H_DIM;   // 786432
  const int N_WS   = E_DIM * H_DIM;       // 524288
  const int N_WH   = E_DIM * E_DIM;       // 1048576
  __bf16* wih1b = Bp; Bp += N_WIH1;
  __bf16* whh1b = Bp; Bp += N_WHH1;
  __bf16* wih2b = Bp; Bp += N_WIH2;
  __bf16* whh2b = Bp; Bp += N_WHH2;
  __bf16* Wsb   = Bp; Bp += N_WS;
  __bf16* Whb   = Bp; Bp += N_WH;

  // ---- one-time work ----
  cvt_bf16_kernel<<<(N_WIH1 + 255) / 256, 256, 0, stream>>>(wih1, wih1b, N_WIH1);
  cvt_bf16_kernel<<<(N_WHH1 + 255) / 256, 256, 0, stream>>>(whh1, whh1b, N_WHH1);
  cvt_bf16_kernel<<<(N_WIH2 + 255) / 256, 256, 0, stream>>>(wih2, wih2b, N_WIH2);
  cvt_bf16_kernel<<<(N_WHH2 + 255) / 256, 256, 0, stream>>>(whh2, whh2b, N_WHH2);
  cvt_bf16_kernel<<<(N_WS   + 255) / 256, 256, 0, stream>>>(W_s,  Wsb,   N_WS);
  cvt_bf16_kernel<<<(N_WH   + 255) / 256, 256, 0, stream>>>(W_h,  Whb,   N_WH);

  init_kernel<<<64, 256, 0, stream>>>(h0, cov0, hidden, cov);
  whenc_kernel<<<dim3(S_DIM * B_DIM / 16, E_DIM / 64), 128, 0, stream>>>(
      enc, Whb, whenc);
  xidx_kernel<<<(T_DIM * B_DIM * S_DIM + 255) / 256, 256, 0, stream>>>(
      xidx, out_xi);

  // ---- sequential scan over T ----
  for (int t = 0; t < T_DIM; ++t) {
    const float* y_t = emb + (size_t)t * B_DIM * I_DIM;
    const float* ym  = ymask + (size_t)t * B_DIM;

    gru_kernel<<<H_DIM / 64, 128, 0, stream>>>(y_t, I_DIM, hidden, wih1b,
                                               whh1b, bih1, bhh1, ym, s1,
                                               nullptr);
    proj_kernel<<<E_DIM / 64, 128, 0, stream>>>(s1, Wsb, battn, proj);
    score_kernel<<<dim3(S_DIM, B_DIM), 128, 0, stream>>>(whenc, proj, cov, W_c,
                                                         Vv, xmask, score);
    softmax_kernel<<<B_DIM, 512, 0, stream>>>(
        score, aw, cov, out_wa + (size_t)t * B_DIM * S_DIM,
        out_cov + (size_t)t * B_DIM * S_DIM);
    ctx_kernel<<<dim3(E_DIM / 256, B_DIM), 256, 0, stream>>>(
        aw, enc, ctx, out_ctx + (size_t)t * B_DIM * E_DIM);
    gru_kernel<<<H_DIM / 64, 128, 0, stream>>>(
        ctx, E_DIM, s1, wih2b, whh2b, bih2, bhh2, ym, hidden,
        out_hid + (size_t)t * B_DIM * H_DIM);
  }
}
